// MultiHeadCrossAttention_15951508537454
// MI455X (gfx1250) — matrix operand order repaired and checked by reading, not compiled, not run on hardware
//
#include <hip/hip_runtime.h>
#include <hip/hip_bf16.h>
#include <math.h>

// ---------------------------------------------------------------------------
// MultiHeadCrossAttention for MI455X (gfx1250, wave32, WMMA).
// B=2, T=S=2048, HIDDEN=1024, H=16, Dh=64.
// All GEMMs use v_wmma_f32_16x16x32_f16 (f16 in, f32 accumulate).
// Double buffering is done by unroll-by-2 ping-pong (no register rotation,
// so no v_mov copies / hazard nops between WMMA groups).
// ---------------------------------------------------------------------------

#define HIDDEN   1024
#define NHEADS   16
#define HEADDIM  64
#define BATCH    2
#define SEQ      2048
#define WAVES    4          // waves per block

typedef _Float16 v16h __attribute__((ext_vector_type(16)));
typedef _Float16 v8h  __attribute__((ext_vector_type(8)));
typedef float    v8f  __attribute__((ext_vector_type(8)));

__device__ __forceinline__ v8f wmma_f16(v16h a, v16h b, v8f c) {
    // 8 args: (neg_a, A, neg_b, B, c_mod, C, reuse_a, reuse_b)
    return __builtin_amdgcn_wmma_f32_16x16x32_f16(false, a, false, b,
                                                  (short)0, c, false, false);
}

// A-fragment: 16x32 f16 tile, row-major source, base points at (row0, k0).
// ISA layout: lanes 0-15 -> M=lane, K = {0..7, 16..23}; lanes 16-31 -> K+8.
__device__ __forceinline__ v16h load_a_frag(const _Float16* base, int lda) {
    const int lane = threadIdx.x & 31;
    const _Float16* p = base + (lane & 15) * lda + ((lane >> 4) << 3);
    v8h lo = *(const v8h*)(p);
    v8h hi = *(const v8h*)(p + 16);
    v16h r;
#pragma unroll
    for (int i = 0; i < 8; ++i) { r[i] = lo[i]; r[i + 8] = hi[i]; }
    return r;
}

// B-fragment: 32x16 f16 tile from COLUMN-contiguous storage (i.e. B^T stored
// row-major with leading dim ldb). base points at element (k0, n0) == BT + n0*ldb + k0.
// ISA layout: lanes 0-15 -> N=lane, K = {0..7, 16..23}; lanes 16-31 -> K+8.
__device__ __forceinline__ v16h load_b_frag(const _Float16* base, int ldb) {
    const int lane = threadIdx.x & 31;
    const _Float16* p = base + (lane & 15) * ldb + ((lane >> 4) << 3);
    v8h lo = *(const v8h*)(p);
    v8h hi = *(const v8h*)(p + 16);
    v16h r;
#pragma unroll
    for (int i = 0; i < 8; ++i) { r[i] = lo[i]; r[i + 8] = hi[i]; }
    return r;
}

// ---------------------------------------------------------------------------
// GEMM helpers: 32 rows x 64 cols per wave (2 M-tiles x 4 N-tiles).
// ---------------------------------------------------------------------------
__device__ __forceinline__ void load_ab(const _Float16* Xb, const _Float16* Wb,
                                        int k, v16h& a0, v16h& a1, v16h bf[4]) {
    a0 = load_a_frag(Xb + k, HIDDEN);
    a1 = load_a_frag(Xb + (size_t)16 * HIDDEN + k, HIDDEN);
#pragma unroll
    for (int nt = 0; nt < 4; ++nt)
        bf[nt] = load_b_frag(Wb + (size_t)(nt * 16) * HIDDEN + k, HIDDEN);
}

__device__ __forceinline__ void mma8(v8f acc[2][4], v16h a0, v16h a1,
                                     const v16h bf[4]) {
#pragma unroll
    for (int nt = 0; nt < 4; ++nt) {
        acc[0][nt] = wmma_f16(a0, bf[nt], acc[0][nt]);
        acc[1][nt] = wmma_f16(a1, bf[nt], acc[1][nt]);
    }
}

// ---------------------------------------------------------------------------
// Packing kernels (run once per launch; tiny vs. GEMM cost)
// ---------------------------------------------------------------------------
__global__ void cvt_f32_to_f16(const float* __restrict__ x,
                               _Float16* __restrict__ y, int n) {
    int i = blockIdx.x * blockDim.x + threadIdx.x;
    if (i < n) y[i] = (_Float16)x[i];
}

// W [H_, K_, N_] f32  ->  WT [H_, N_, K_] f16, scaled.
__global__ void pack_wT(const float* __restrict__ W, _Float16* __restrict__ WT,
                        float scale, int K_, int N_, int total) {
    int i = blockIdx.x * blockDim.x + threadIdx.x;
    if (i >= total) return;
    int h   = i / (K_ * N_);
    int rem = i - h * (K_ * N_);
    int k   = rem / N_;
    int n   = rem - k * N_;
    WT[(h * N_ + n) * K_ + k] = (_Float16)(W[i] * scale);
}

// ---------------------------------------------------------------------------
// Per-head projection GEMM: Y[b,h] = X[b] (SEQ x 1024) @ W[h] (1024 x 64) + bias
// Ping-pong double buffering (unroll-by-2, branch-free inner loop).
// transpose_store==1 stores Y^T as [B,H,64,SEQ] (used for V).
// ---------------------------------------------------------------------------
__global__ __launch_bounds__(WAVES * 32)
void proj_kernel(const _Float16* __restrict__ X,   // [B, SEQ, 1024] f16
                 const _Float16* __restrict__ WT,  // [H, 64, 1024]  f16 (pre-scaled)
                 const float*    __restrict__ bias,// [H, 64] f32
                 _Float16* __restrict__ Y,
                 float bias_scale, int transpose_store) {
    const int lane = threadIdx.x & 31;
    const int wave = threadIdx.x >> 5;
    const int h = blockIdx.y;
    const int b = blockIdx.z;
    const int t0 = (blockIdx.x * WAVES + wave) * 32;   // 32 rows per wave

    const _Float16* Xb  = X + (size_t)b * SEQ * HIDDEN + (size_t)t0 * HIDDEN;
    const _Float16* WTh = WT + (size_t)h * HEADDIM * HIDDEN;

    v8f acc[2][4];
#pragma unroll
    for (int mt = 0; mt < 2; ++mt)
#pragma unroll
        for (int nt = 0; nt < 4; ++nt) acc[mt][nt] = (v8f)(0.0f);

    v16h aA0, aA1, bfA[4];          // ping buffer
    v16h aB0, aB1, bfB[4];          // pong buffer
    load_ab(Xb, WTh, 0, aA0, aA1, bfA);

    int k0 = 0;
    for (; k0 + 64 < HIDDEN; k0 += 64) {
        load_ab(Xb, WTh, k0 + 32, aB0, aB1, bfB);   // in flight during A-phase
        mma8(acc, aA0, aA1, bfA);
        load_ab(Xb, WTh, k0 + 64, aA0, aA1, bfA);   // in flight during B-phase
        mma8(acc, aB0, aB1, bfB);
    }
    // tail: k0 == HIDDEN-64
    load_ab(Xb, WTh, k0 + 32, aB0, aB1, bfB);
    mma8(acc, aA0, aA1, bfA);
    mma8(acc, aB0, aB1, bfB);

    const int n   = lane & 15;
    const int mhi = (lane >> 4) << 3;   // +8 rows for upper half-wave
#pragma unroll
    for (int nt = 0; nt < 4; ++nt) {
        float bv = bias[h * HEADDIM + nt * 16 + n] * bias_scale;
#pragma unroll
        for (int mt = 0; mt < 2; ++mt)
#pragma unroll
            for (int r = 0; r < 8; ++r) acc[mt][nt][r] += bv;
    }

    if (!transpose_store) {
        // Y [B,H,SEQ,64] row-major
        _Float16* Yh = Y + ((size_t)(b * NHEADS + h) * SEQ + t0) * HEADDIM;
#pragma unroll
        for (int mt = 0; mt < 2; ++mt)
#pragma unroll
            for (int nt = 0; nt < 4; ++nt)
#pragma unroll
                for (int r = 0; r < 8; ++r)
                    Yh[(size_t)(mt * 16 + r + mhi) * HEADDIM + nt * 16 + n] =
                        (_Float16)acc[mt][nt][r];
    } else {
        // Y^T [B,H,64,SEQ]: element (d, s)
        _Float16* Yh = Y + (size_t)(b * NHEADS + h) * HEADDIM * SEQ;
#pragma unroll
        for (int mt = 0; mt < 2; ++mt)
#pragma unroll
            for (int nt = 0; nt < 4; ++nt)
#pragma unroll
                for (int r = 0; r < 8; ++r)
                    Yh[(size_t)(nt * 16 + n) * SEQ + t0 + mt * 16 + r + mhi] =
                        (_Float16)acc[mt][nt][r];
    }
}

// ---------------------------------------------------------------------------
// Flash attention step for one 32-column S-tile:
//   V frags issued first, 4 score WMMAs with kb, next K tile loaded into kbn
//   (overlaps softmax VALU), online softmax, P re-layout C->A via per-wave
//   LDS patch, 4 P@V WMMAs. kb/kbn alternate across calls (ping-pong).
// ---------------------------------------------------------------------------
__device__ __forceinline__ void attn_step(
    v16h qa0, v16h qa1,
    const v16h kb[4], v16h kbn[4],
    const _Float16* Kn,            // base of next K tile, or nullptr
    const _Float16* VTh, int s0,
    v8f O[4], float mrow[8], float lrow[8],
    _Float16* myP, int n, int mhi, int lane) {

    // V fragment loads (independent of score math; V^T column-contiguous)
    v16h vb[4];
#pragma unroll
    for (int d = 0; d < 4; ++d)
        vb[d] = load_b_frag(VTh + (size_t)(d * 16) * SEQ + s0, SEQ);

    // scores: 16 x 32 tile (t x s), log2-domain
    v8f sc0 = (v8f)(0.0f), sc1 = (v8f)(0.0f);
    sc0 = wmma_f16(qa0, kb[0], sc0);
    sc0 = wmma_f16(qa1, kb[1], sc0);
    sc1 = wmma_f16(qa0, kb[2], sc1);
    sc1 = wmma_f16(qa1, kb[3], sc1);

    // next K tile fragments; overlap with softmax VALU tail
    if (Kn) {
        kbn[0] = load_b_frag(Kn + 0,                 HEADDIM);
        kbn[1] = load_b_frag(Kn + 32,                HEADDIM);
        kbn[2] = load_b_frag(Kn + 16 * HEADDIM,      HEADDIM);
        kbn[3] = load_b_frag(Kn + 16 * HEADDIM + 32, HEADDIM);
        // warm caches one tile further ahead (global_prefetch_b8)
        __builtin_prefetch(Kn + 32 * HEADDIM + lane * HEADDIM, 0, 1);
        __builtin_prefetch(VTh + (size_t)lane * SEQ + s0 + 64, 0, 1);
    }

    // row max over the 32 new columns (cross 16 lanes of half-wave)
    float mn[8];
#pragma unroll
    for (int r = 0; r < 8; ++r) mn[r] = fmaxf(sc0[r], sc1[r]);
#pragma unroll
    for (int off = 1; off < 16; off <<= 1)
#pragma unroll
        for (int r = 0; r < 8; ++r)
            mn[r] = fmaxf(mn[r], __shfl_xor(mn[r], off, 32));

    float scale[8], rs[8];
#pragma unroll
    for (int r = 0; r < 8; ++r) {
        float mnew = fmaxf(mrow[r], mn[r]);
        scale[r] = exp2f(mrow[r] - mnew);   // 0 on first step (exp2(-inf))
        mrow[r] = mnew;
    }
    // P = exp2(s - m), row sums
#pragma unroll
    for (int r = 0; r < 8; ++r) {
        float p0 = exp2f(sc0[r] - mrow[r]);
        float p1 = exp2f(sc1[r] - mrow[r]);
        sc0[r] = p0; sc1[r] = p1; rs[r] = p0 + p1;
    }
#pragma unroll
    for (int off = 1; off < 16; off <<= 1)
#pragma unroll
        for (int r = 0; r < 8; ++r)
            rs[r] += __shfl_xor(rs[r], off, 32);
#pragma unroll
    for (int r = 0; r < 8; ++r) lrow[r] = lrow[r] * scale[r] + rs[r];

    // rescale running O
#pragma unroll
    for (int d = 0; d < 4; ++d)
#pragma unroll
        for (int r = 0; r < 8; ++r) O[d][r] *= scale[r];

    // re-layout P (C layout -> A layout) through per-wave LDS
#pragma unroll
    for (int r = 0; r < 8; ++r) {
        myP[(r + mhi) * 32 + n]      = (_Float16)sc0[r];
        myP[(r + mhi) * 32 + 16 + n] = (_Float16)sc1[r];
    }
    // Same-wave DS ops are in-order; compiler inserts s_wait_dscnt for reads.
    v16h pa = load_a_frag(myP, 32);

    // O += P @ V
#pragma unroll
    for (int d = 0; d < 4; ++d) O[d] = wmma_f16(pa, vb[d], O[d]);
}

__global__ __launch_bounds__(WAVES * 32)
void attn_kernel(const _Float16* __restrict__ Q,   // [B,H,SEQ,64]  (pre-scaled)
                 const _Float16* __restrict__ K,   // [B,H,SEQ,64]
                 const _Float16* __restrict__ VT,  // [B,H,64,SEQ]
                 _Float16* __restrict__ Hc) {      // [B,SEQ,1024] concat heads
    const int lane = threadIdx.x & 31;
    const int wave = threadIdx.x >> 5;
    const int h = blockIdx.y;
    const int b = blockIdx.z;
    const int t0 = (blockIdx.x * WAVES + wave) * 16;

    const _Float16* Qh  = Q  + ((size_t)(b * NHEADS + h) * SEQ + t0) * HEADDIM;
    const _Float16* Kh  = K  + (size_t)(b * NHEADS + h) * SEQ * HEADDIM;
    const _Float16* VTh = VT + (size_t)(b * NHEADS + h) * HEADDIM * SEQ;

    // Loop-invariant Q A-fragments (Dh=64 -> two K=32 fragments)
    v16h qa0 = load_a_frag(Qh + 0,  HEADDIM);
    v16h qa1 = load_a_frag(Qh + 32, HEADDIM);

    __shared__ _Float16 Pst[WAVES][16 * 32];
    _Float16* myP = &Pst[wave][0];

    v8f O[4];
#pragma unroll
    for (int d = 0; d < 4; ++d) O[d] = (v8f)(0.0f);
    float mrow[8], lrow[8];
#pragma unroll
    for (int r = 0; r < 8; ++r) { mrow[r] = -INFINITY; lrow[r] = 0.0f; }

    const int n   = lane & 15;
    const int mhi = (lane >> 4) << 3;

    // ping-pong K fragment buffers
    v16h kbA[4], kbB[4];
    kbA[0] = load_b_frag(Kh + 0,                 HEADDIM);
    kbA[1] = load_b_frag(Kh + 32,                HEADDIM);
    kbA[2] = load_b_frag(Kh + 16 * HEADDIM,      HEADDIM);
    kbA[3] = load_b_frag(Kh + 16 * HEADDIM + 32, HEADDIM);

    int s0 = 0;
    for (; s0 + 64 < SEQ; s0 += 64) {
        attn_step(qa0, qa1, kbA, kbB, Kh + (size_t)(s0 + 32) * HEADDIM,
                  VTh, s0,      O, mrow, lrow, myP, n, mhi, lane);
        attn_step(qa0, qa1, kbB, kbA, Kh + (size_t)(s0 + 64) * HEADDIM,
                  VTh, s0 + 32, O, mrow, lrow, myP, n, mhi, lane);
    }
    // tail pair: s0 == SEQ-64
    attn_step(qa0, qa1, kbA, kbB, Kh + (size_t)(s0 + 32) * HEADDIM,
              VTh, s0,      O, mrow, lrow, myP, n, mhi, lane);
    attn_step(qa0, qa1, kbB, kbA, (const _Float16*)nullptr,
              VTh, s0 + 32, O, mrow, lrow, myP, n, mhi, lane);

    // epilogue: O /= l, store concat-head f16 [B,SEQ,1024]
    float linv[8];
#pragma unroll
    for (int r = 0; r < 8; ++r) linv[r] = 1.0f / lrow[r];
#pragma unroll
    for (int d = 0; d < 4; ++d)
#pragma unroll
        for (int r = 0; r < 8; ++r) {
            float o = O[d][r] * linv[r];
            Hc[((size_t)(b * SEQ + t0 + r + mhi)) * HIDDEN
               + h * HEADDIM + d * 16 + n] = (_Float16)o;
        }
}

// ---------------------------------------------------------------------------
// Output projection: out = Hc (4096 x 1024) @ Wo + bo, f32 output.
// One wave -> 32 rows x 64 cols, ping-pong double buffering.
// ---------------------------------------------------------------------------
__global__ __launch_bounds__(WAVES * 32)
void out_proj_kernel(const _Float16* __restrict__ Hc,   // [B*SEQ, 1024]
                     const _Float16* __restrict__ WoT,  // [1024(n), 1024(k)]
                     const float*    __restrict__ bo,   // [1024]
                     float* __restrict__ out) {         // [B*SEQ, 1024]
    const int lane = threadIdx.x & 31;
    const int wave = threadIdx.x >> 5;
    const int row0 = (blockIdx.x * WAVES + wave) * 32;
    const int nc0  = blockIdx.y * 64;

    const _Float16* A  = Hc + (size_t)row0 * HIDDEN;
    const _Float16* Bc = WoT + (size_t)nc0 * HIDDEN;

    v8f acc[2][4];
#pragma unroll
    for (int mt = 0; mt < 2; ++mt)
#pragma unroll
        for (int nt = 0; nt < 4; ++nt) acc[mt][nt] = (v8f)(0.0f);

    v16h aA0, aA1, bfA[4];
    v16h aB0, aB1, bfB[4];
    load_ab(A, Bc, 0, aA0, aA1, bfA);

    int k0 = 0;
    for (; k0 + 64 < HIDDEN; k0 += 64) {
        load_ab(A, Bc, k0 + 32, aB0, aB1, bfB);
        mma8(acc, aA0, aA1, bfA);
        load_ab(A, Bc, k0 + 64, aA0, aA1, bfA);
        mma8(acc, aB0, aB1, bfB);
    }
    load_ab(A, Bc, k0 + 32, aB0, aB1, bfB);
    mma8(acc, aA0, aA1, bfA);
    mma8(acc, aB0, aB1, bfB);

    const int n   = lane & 15;
    const int mhi = (lane >> 4) << 3;
#pragma unroll
    for (int nt = 0; nt < 4; ++nt) {
        float bv = bo[nc0 + nt * 16 + n];
#pragma unroll
        for (int mt = 0; mt < 2; ++mt)
#pragma unroll
            for (int r = 0; r < 8; ++r)
                out[(size_t)(row0 + mt * 16 + r + mhi) * HIDDEN + nc0 + nt * 16 + n] =
                    acc[mt][nt][r] + bv;
    }
}

// ---------------------------------------------------------------------------
// Host launcher
// ---------------------------------------------------------------------------
extern "C" void kernel_launch(void* const* d_in, const int* in_sizes, int n_in,
                              void* d_out, int out_size, void* d_ws, size_t ws_size,
                              hipStream_t stream) {
    const float* tgt = (const float*)d_in[0];  // [B,SEQ,1024]
    const float* src = (const float*)d_in[1];  // [B,SEQ,1024]
    const float* Wq  = (const float*)d_in[2];  // [H,1024,64]
    const float* bq  = (const float*)d_in[3];  // [H,64]
    const float* Wk  = (const float*)d_in[4];
    const float* bk  = (const float*)d_in[5];
    const float* Wv  = (const float*)d_in[6];
    const float* bv  = (const float*)d_in[7];
    const float* Wo  = (const float*)d_in[8];  // [1024,1024]
    const float* bo  = (const float*)d_in[9];  // [1024]
    float* out = (float*)d_out;

    // Workspace layout (f16 tensors), all offsets 256B-aligned.
    char* ws = (char*)d_ws;
    size_t off = 0;
    auto alloc = [&](size_t bytes) {
        char* p = ws + off;
        off += (bytes + 255) & ~(size_t)255;
        return p;
    };
    const size_t actH = (size_t)BATCH * SEQ * HIDDEN;          // 4 M elems
    const size_t wHKN = (size_t)NHEADS * HIDDEN * HEADDIM;     // 1 M elems
    const size_t qkv  = (size_t)BATCH * NHEADS * SEQ * HEADDIM;

    _Float16* tgtH = (_Float16*)alloc(actH * 2);
    _Float16* srcH = (_Float16*)alloc(actH * 2);
    _Float16* WqT  = (_Float16*)alloc(wHKN * 2);
    _Float16* WkT  = (_Float16*)alloc(wHKN * 2);
    _Float16* WvT  = (_Float16*)alloc(wHKN * 2);
    _Float16* WoT  = (_Float16*)alloc((size_t)HIDDEN * HIDDEN * 2);
    _Float16* Qf   = (_Float16*)alloc(qkv * 2);
    _Float16* Kf   = (_Float16*)alloc(qkv * 2);
    _Float16* VTf  = (_Float16*)alloc(qkv * 2);
    _Float16* Hc   = (_Float16*)alloc(actH * 2);
    (void)ws_size; (void)n_in; (void)in_sizes; (void)out_size;

    // Fold softmax scale and log2(e) into Q path so attention uses exp2.
    const float qscale = (1.0f / 8.0f) * 1.44269504088896340736f; // 1/sqrt(64)*log2e

    // 1) pack / convert
    cvt_f32_to_f16<<<dim3((unsigned)((actH + 255) / 256)), dim3(256), 0, stream>>>(tgt, tgtH, (int)actH);
    cvt_f32_to_f16<<<dim3((unsigned)((actH + 255) / 256)), dim3(256), 0, stream>>>(src, srcH, (int)actH);
    pack_wT<<<dim3((unsigned)((wHKN + 255) / 256)), dim3(256), 0, stream>>>(Wq, WqT, qscale, HIDDEN, HEADDIM, (int)wHKN);
    pack_wT<<<dim3((unsigned)((wHKN + 255) / 256)), dim3(256), 0, stream>>>(Wk, WkT, 1.0f,  HIDDEN, HEADDIM, (int)wHKN);
    pack_wT<<<dim3((unsigned)((wHKN + 255) / 256)), dim3(256), 0, stream>>>(Wv, WvT, 1.0f,  HIDDEN, HEADDIM, (int)wHKN);
    pack_wT<<<dim3((unsigned)((HIDDEN * HIDDEN + 255) / 256)), dim3(256), 0, stream>>>(Wo, WoT, 1.0f, HIDDEN, HIDDEN, HIDDEN * HIDDEN);

    // 2) projections (Q scaled; V stored transposed for contiguous B-frags)
    dim3 pblk(WAVES * 32);
    dim3 pgrid(SEQ / (32 * WAVES), NHEADS, BATCH);   // 32 rows per wave
    proj_kernel<<<pgrid, pblk, 0, stream>>>(tgtH, WqT, bq, Qf,  qscale, 0);
    proj_kernel<<<pgrid, pblk, 0, stream>>>(srcH, WkT, bk, Kf,  1.0f,   0);
    proj_kernel<<<pgrid, pblk, 0, stream>>>(srcH, WvT, bv, VTf, 1.0f,   1);

    // 3) flash attention -> concat heads (f16); 16 rows per wave
    dim3 agrid(SEQ / (16 * WAVES), NHEADS, BATCH);
    attn_kernel<<<agrid, pblk, 0, stream>>>(Qf, Kf, VTf, Hc);

    // 4) output projection (f32 out); 32 rows per wave
    dim3 ogrid((BATCH * SEQ) / (32 * WAVES), HIDDEN / 64, 1);
    out_proj_kernel<<<ogrid, pblk, 0, stream>>>(Hc, WoT, bo, out);
}
